// rFFTPooling2D_20194936226429
// MI455X (gfx1250) — matrix operand order repaired and checked by reading, not compiled, not run on hardware
//
#include <hip/hip_runtime.h>

// rFFT pooling: y = irfft2( rfft2(x)[:, :, 128:256, 0:65] )  per (b,c) image.
// Factored into two real GEMM stages with on-device-built DFT weight matrices.
// Stage A: z(r,i)[b, m, w*c]   = WA(r,i)[128x256] . x[b, h, w*c]      (N = 24576)
// Stage B: y[b, m, n, c]       = P[128x256] . zr  + Q[128x256] . zi   (per (b,m), N = 96)
// bf16 operands, f32 WMMA accumulate; memory-bound (~0.9 GB -> ~39 us @ 23.3 TB/s),
// so staging is vectorized + register-pipelined + LDS double-buffered.

typedef __attribute__((ext_vector_type(16))) __bf16 v16bf;
typedef __attribute__((ext_vector_type(8)))  __bf16 v8bf;
typedef __attribute__((ext_vector_type(8)))  float  v8f;

#define LDS_PITCH 40   // k-tile 32 + 8 pad (bf16 elems); row stride 80B keeps 16B alignment

__device__ __forceinline__ v16bf cat16(v8bf lo, v8bf hi) {
  return __builtin_shufflevector(lo, hi, 0,1,2,3,4,5,6,7,8,9,10,11,12,13,14,15);
}

// A fragment (16x32 bf16): lanes 0-15 = rows, lanes 16-31 = same rows, upper-K halves.
__device__ __forceinline__ v16bf load_a_frag(const __bf16* __restrict__ W, int mbase, int k0, int lane) {
  const int row  = mbase + (lane & 15);
  const int half = lane >> 4;
  const __bf16* p = W + row * 256 + k0 + half * 8;
  v8bf lo = *(const v8bf*)(p);
  v8bf hi = *(const v8bf*)(p + 16);
  return cat16(lo, hi);
}

// B fragment (32x16 bf16) from LDS staged transposed: LDS[n][k], pitch LDS_PITCH.
__device__ __forceinline__ v16bf load_b_frag(const __bf16* lds, int nbase, int lane) {
  const int n    = nbase + (lane & 15);
  const int half = lane >> 4;
  const __bf16* p = lds + n * LDS_PITCH + half * 8;
  v8bf lo = *(const v8bf*)(p);
  v8bf hi = *(const v8bf*)(p + 16);
  return cat16(lo, hi);
}

#define WMMA_BF16(a, b, c) \
  __builtin_amdgcn_wmma_f32_16x16x32_bf16(false, (a), false, (b), (short)0, (c), false, false)

__device__ __forceinline__ unsigned int pack2_f32_bf16(float lo, float hi) {
  unsigned short ul = __builtin_bit_cast(unsigned short, (__bf16)lo);
  unsigned short uh = __builtin_bit_cast(unsigned short, (__bf16)hi);
  return (unsigned int)ul | ((unsigned int)uh << 16);
}

// Store a 2(k) x 4(n) micro-tile transposed into LDS as 4x ds_store_b32.
__device__ __forceinline__ void stage_tile_f32(__bf16* lds, int kk, int nn,
                                               const float4& r0, const float4& r1) {
  const float* a = &r0.x;
  const float* b = &r1.x;
  #pragma unroll
  for (int j = 0; j < 4; ++j) {
    *(unsigned int*)((unsigned short*)lds + (nn + j) * LDS_PITCH + kk) = pack2_f32_bf16(a[j], b[j]);
  }
}

__device__ __forceinline__ void stage_tile_bf16(__bf16* lds, int kk, int nn,
                                                const ushort4& r0, const ushort4& r1) {
  const unsigned short* a = &r0.x;
  const unsigned short* b = &r1.x;
  #pragma unroll
  for (int j = 0; j < 4; ++j) {
    *(unsigned int*)((unsigned short*)lds + (nn + j) * LDS_PITCH + kk) =
        (unsigned int)a[j] | ((unsigned int)b[j] << 16);
  }
}

// ---------------------------------------------------------------------------
// Weight construction (exact operator algebra, evaluated in f32, stored bf16)
// ---------------------------------------------------------------------------
__global__ __launch_bounds__(256) void init_weights_kernel(
    __bf16* __restrict__ WAr, __bf16* __restrict__ WAi,
    __bf16* __restrict__ WP,  __bf16* __restrict__ WQ) {
  const int m   = blockIdx.x;    // 0..127
  const int col = threadIdx.x;   // 0..255
  const float PI = 3.14159265358979323846f;

  // Row operator: z[m] = x[2m] - (1/128) sum_{h odd} x[h] - (i/128) sum_{h odd} cot(pi(2m-h)/256) x[h]
  float war = (col == 2 * m ? 1.0f : 0.0f) - ((col & 1) ? (1.0f / 128.0f) : 0.0f);
  float wai = 0.0f;
  if (col & 1) {
    float t = PI * (float)(2 * m - col) / 256.0f;
    wai = -(1.0f / 128.0f) * (cosf(t) / sinf(t));
  }
  WAr[m * 256 + col] = (__bf16)war;
  WAi[m * 256 + col] = (__bf16)wai;

  // Column operator (irfft over 65 retained bins, Re at DC/Nyquist bins):
  // P = (1/128)[1 + 2 sum_{q=1}^{63} cos(pi q (w-2n)/128) + (-1)^n cos(pi w/2)]
  // Q = (1/128)[    2 sum_{q=1}^{63} sin(pi q (w-2n)/128) + (-1)^n sin(pi w/2)]
  const int d = col - 2 * m;
  float sP = 1.0f, sQ = 0.0f;
  for (int q = 1; q < 64; ++q) {
    int qd = ((q * d) % 256 + 256) & 255;   // exact range reduction (period 256)
    float a = PI * (float)qd / 128.0f;
    sP += 2.0f * cosf(a);
    sQ += 2.0f * sinf(a);
  }
  const float sgn = (m & 1) ? -1.0f : 1.0f;
  const int   w4  = col & 3;
  sP += sgn * ((w4 == 0) ? 1.0f : ((w4 == 2) ? -1.0f : 0.0f));
  sQ += sgn * ((w4 == 1) ? 1.0f : ((w4 == 3) ? -1.0f : 0.0f));
  WP[m * 256 + col] = (__bf16)(sP / 128.0f);
  WQ[m * 256 + col] = (__bf16)(sQ / 128.0f);
}

// ---------------------------------------------------------------------------
// Stage A: per b, [128x256] x [256x24576] -> zr, zi (two outputs, shared B tile)
// 8 waves, tile M=128 x N=64; wave = (m-quad, n-half): 2 m-frags x 2 n-frags x 2 mats.
// Register-pipelined staging, double-buffered LDS, one barrier per k-iter.
// ---------------------------------------------------------------------------
__global__ __launch_bounds__(256) void stageA_kernel(
    const float* __restrict__ x,
    const __bf16* __restrict__ WAr, const __bf16* __restrict__ WAi,
    __bf16* __restrict__ zr, __bf16* __restrict__ zi) {
  __shared__ __align__(16) __bf16 ldsB[2][64 * LDS_PITCH];
  constexpr int NTOT = 256 * 96;   // 24576

  const int b    = blockIdx.y;
  const int n0   = blockIdx.x * 64;
  const int tid  = threadIdx.x;
  const int lane = tid & 31;
  const int wave = tid >> 5;
  const int m0   = (wave >> 1) * 32;
  const int nh   = (wave & 1) * 32;
  const size_t xbase = (size_t)b * 256 * NTOT;
  const size_t zbase = (size_t)b * 128 * NTOT;

  // Staging micro-tile owned by this thread: rows (kk, kk+1), cols nn..nn+3.
  const int kk = (tid >> 4) * 2;
  const int nn = (tid & 15) * 4;
  const float* xp = x + xbase + n0 + nn;

  v8f accr[2][2] = {{{0.f},{0.f}},{{0.f},{0.f}}};
  v8f acci[2][2] = {{{0.f},{0.f}},{{0.f},{0.f}}};

  // Prologue: stage tile k0 = 0 into buffer 0.
  float4 r0 = *(const float4*)(xp + (size_t)(kk)     * NTOT);
  float4 r1 = *(const float4*)(xp + (size_t)(kk + 1) * NTOT);
  stage_tile_f32(ldsB[0], kk, nn, r0, r1);
  __syncthreads();

  for (int k0 = 0; k0 < 256; k0 += 32) {
    const int  cur  = (k0 >> 5) & 1;
    const bool more = (k0 + 32) < 256;

    // Issue next tile's global loads early (latency hidden under WMMA).
    if (more) {
      r0 = *(const float4*)(xp + (size_t)(k0 + 32 + kk)     * NTOT);
      r1 = *(const float4*)(xp + (size_t)(k0 + 32 + kk + 1) * NTOT);
    }
    if (k0 + 64 < 256)  // global_prefetch_b8 for the +2 tile
      __builtin_prefetch(xp + (size_t)(k0 + 64 + kk) * NTOT, 0, 1);

    v16bf bf0 = load_b_frag(ldsB[cur], nh +  0, lane);
    v16bf bf1 = load_b_frag(ldsB[cur], nh + 16, lane);
    #pragma unroll
    for (int mi = 0; mi < 2; ++mi) {
      v16bf ar = load_a_frag(WAr, m0 + mi * 16, k0, lane);
      v16bf ai = load_a_frag(WAi, m0 + mi * 16, k0, lane);
      accr[mi][0] = WMMA_BF16(ar, bf0, accr[mi][0]);
      accr[mi][1] = WMMA_BF16(ar, bf1, accr[mi][1]);
      acci[mi][0] = WMMA_BF16(ai, bf0, acci[mi][0]);
      acci[mi][1] = WMMA_BF16(ai, bf1, acci[mi][1]);
    }

    if (more) stage_tile_f32(ldsB[cur ^ 1], kk, nn, r0, r1);
    __syncthreads();
  }

  // C/D layout: VGPR r -> (M = r + 8*half, N = lane%16)
  const int half = lane >> 4, nl = lane & 15;
  #pragma unroll
  for (int mi = 0; mi < 2; ++mi)
    #pragma unroll
    for (int ni = 0; ni < 2; ++ni)
      #pragma unroll
      for (int r = 0; r < 8; ++r) {
        int row = m0 + mi * 16 + r + 8 * half;
        int col = n0 + nh + ni * 16 + nl;
        size_t o = zbase + (size_t)row * NTOT + col;
        zr[o] = (__bf16)accr[mi][ni][r];
        zi[o] = (__bf16)acci[mi][ni][r];
      }
}

// ---------------------------------------------------------------------------
// Stage B: per (b,m), y[128x96] = P.zr_bm + Q.zi_bm   (K = 256 w, N = 96 c)
// 12 waves; wave = (m-pair, n-pair): 2 m-frags x 2 n-frags, two products per acc.
// 384 threads = exactly one staging micro-tile (2w x 4c, both matrices) each.
// ---------------------------------------------------------------------------
__global__ __launch_bounds__(384) void stageB_kernel(
    const __bf16* __restrict__ zr, const __bf16* __restrict__ zi,
    const __bf16* __restrict__ WP, const __bf16* __restrict__ WQ,
    float* __restrict__ out) {
  __shared__ __align__(16) __bf16 ldsR[2][96 * LDS_PITCH];
  __shared__ __align__(16) __bf16 ldsI[2][96 * LDS_PITCH];

  const int b    = blockIdx.y;
  const int m    = blockIdx.x;          // output h' index
  const int tid  = threadIdx.x;
  const int lane = tid & 31;
  const int wave = tid >> 5;            // 0..11
  const int r0t  = (wave / 3) * 32;     // output w' rows for this wave
  const int c0   = (wave % 3) * 32;     // channel cols for this wave
  const size_t zb = ((size_t)(b * 128 + m)) * 256 * 96;
  const size_t ob = ((size_t)(b * 128 + m)) * 128 * 96;

  // Staging micro-tile: rows (kk, kk+1) of w, cols nn..nn+3 of c.
  const int kk = (tid / 24) * 2;
  const int nn = (tid % 24) * 4;
  const __bf16* zrp = zr + zb + nn;
  const __bf16* zip = zi + zb + nn;

  v8f acc[2][2] = {{{0.f},{0.f}},{{0.f},{0.f}}};

  // Prologue: stage tile k0 = 0.
  ushort4 a0 = *(const ushort4*)(zrp + (size_t)(kk)     * 96);
  ushort4 a1 = *(const ushort4*)(zrp + (size_t)(kk + 1) * 96);
  ushort4 b0 = *(const ushort4*)(zip + (size_t)(kk)     * 96);
  ushort4 b1 = *(const ushort4*)(zip + (size_t)(kk + 1) * 96);
  stage_tile_bf16(ldsR[0], kk, nn, a0, a1);
  stage_tile_bf16(ldsI[0], kk, nn, b0, b1);
  __syncthreads();

  for (int k0 = 0; k0 < 256; k0 += 32) {
    const int  cur  = (k0 >> 5) & 1;
    const bool more = (k0 + 32) < 256;

    if (more) {
      a0 = *(const ushort4*)(zrp + (size_t)(k0 + 32 + kk)     * 96);
      a1 = *(const ushort4*)(zrp + (size_t)(k0 + 32 + kk + 1) * 96);
      b0 = *(const ushort4*)(zip + (size_t)(k0 + 32 + kk)     * 96);
      b1 = *(const ushort4*)(zip + (size_t)(k0 + 32 + kk + 1) * 96);
    }

    v16bf bR0 = load_b_frag(ldsR[cur], c0 +  0, lane);
    v16bf bR1 = load_b_frag(ldsR[cur], c0 + 16, lane);
    v16bf bI0 = load_b_frag(ldsI[cur], c0 +  0, lane);
    v16bf bI1 = load_b_frag(ldsI[cur], c0 + 16, lane);
    #pragma unroll
    for (int mi = 0; mi < 2; ++mi) {
      v16bf aP = load_a_frag(WP, r0t + mi * 16, k0, lane);
      v16bf aQ = load_a_frag(WQ, r0t + mi * 16, k0, lane);
      acc[mi][0] = WMMA_BF16(aP, bR0, acc[mi][0]);
      acc[mi][1] = WMMA_BF16(aP, bR1, acc[mi][1]);
      acc[mi][0] = WMMA_BF16(aQ, bI0, acc[mi][0]);
      acc[mi][1] = WMMA_BF16(aQ, bI1, acc[mi][1]);
    }

    if (more) {
      stage_tile_bf16(ldsR[cur ^ 1], kk, nn, a0, a1);
      stage_tile_bf16(ldsI[cur ^ 1], kk, nn, b0, b1);
    }
    __syncthreads();
  }

  const int half = lane >> 4, nl = lane & 15;
  #pragma unroll
  for (int mi = 0; mi < 2; ++mi)
    #pragma unroll
    for (int ni = 0; ni < 2; ++ni)
      #pragma unroll
      for (int r = 0; r < 8; ++r) {
        int row = r0t + mi * 16 + r + 8 * half;
        int c   = c0 + ni * 16 + nl;
        out[ob + (size_t)row * 96 + c] = acc[mi][ni][r];
      }
}

// ---------------------------------------------------------------------------
extern "C" void kernel_launch(void* const* d_in, const int* in_sizes, int n_in,
                              void* d_out, int out_size, void* d_ws, size_t ws_size,
                              hipStream_t stream) {
  (void)in_sizes; (void)n_in; (void)out_size; (void)ws_size;
  const float* x   = (const float*)d_in[0];
  float*       out = (float*)d_out;

  // Workspace layout (bf16 elems): 4 weight matrices (128x256 each) + zr + zi.
  __bf16* wsb = (__bf16*)d_ws;
  __bf16* WAr = wsb;
  __bf16* WAi = WAr + 128 * 256;
  __bf16* WP  = WAi + 128 * 256;
  __bf16* WQ  = WP  + 128 * 256;
  const size_t ZELEMS = (size_t)16 * 128 * 256 * 96;   // 50,331,648 per matrix
  __bf16* zr = WQ + 128 * 256;
  __bf16* zi = zr + ZELEMS;

  init_weights_kernel<<<dim3(128), dim3(256), 0, stream>>>(WAr, WAi, WP, WQ);
  stageA_kernel<<<dim3(24576 / 64, 16), dim3(256), 0, stream>>>(x, WAr, WAi, zr, zi);
  stageB_kernel<<<dim3(128, 16), dim3(384), 0, stream>>>(zr, zi, WP, WQ, out);
}
